// VisionMamba_11493332484857
// MI455X (gfx1250) — matrix operand order repaired and checked by reading, not compiled, not run on hardware
//
#include <hip/hip_runtime.h>
#include <math.h>

#define L_TOK 1569
#define L_PAD 1600            // 25 groups of 64 rows
#define PM    784             // patch GEMM rows
#define PM_PAD 832            // 13 groups of 64 rows
#define DM    384
#define DI    768
#define DS    16
#define DTR   24
#define DXPP  64              // padded x_proj out dim: [0:24)=dt [24:32)=0 [32:48)=B [48:64)=C
#define KDT   32              // padded dt_proj K
#define DEPTH 24
#define TSEQ  4
#define NPATCH 196
#define SCH   32              // scan LDS chunk (timesteps)

// Tensor Data Mover availability (device pass only; arity differs by toolchain)
#if defined(__gfx1250__)
#  if __has_builtin(__builtin_amdgcn_tensor_load_to_lds)
#    define USE_TDM 1
#  endif
#endif
#ifndef USE_TDM
#  define USE_TDM 0
#endif

typedef __attribute__((ext_vector_type(16))) _Float16 v16h;
typedef __attribute__((ext_vector_type(8)))  _Float16 h8;
typedef __attribute__((ext_vector_type(4)))  _Float16 h4;
typedef __attribute__((ext_vector_type(4)))  float    f4;
typedef __attribute__((ext_vector_type(8)))  float    v8f;
typedef __attribute__((ext_vector_type(4)))  unsigned int u32x4;
typedef __attribute__((ext_vector_type(4)))  int      i32x4v;
typedef __attribute__((ext_vector_type(8)))  int      i32x8v;

__device__ __forceinline__ float dev_silu(float x)     { return x / (1.f + expf(-x)); }
__device__ __forceinline__ float dev_softplus(float x) { return x > 20.f ? x : log1pf(expf(x)); }

__device__ __forceinline__ v16h cat16(h8 lo, h8 hi) {
    return __builtin_shufflevector(lo, hi, 0,1,2,3,4,5,6,7,8,9,10,11,12,13,14,15);
}
// B operand (32x16): lane l -> col N=l%16; halves 0-7 -> K=k0+hi*8+0..7,
// halves 8-15 -> K=k0+16+hi*8+0..7
__device__ __forceinline__ v16h ldB(const _Float16* __restrict__ Bp, int k0, int hi) {
    const _Float16* p = Bp + k0 + hi * 8;
    return cat16(*(const h8*)p, *(const h8*)(p + 16));
}
// A operand (16x32): lane l -> row M=l%16; halves 0-7 -> K=k0+hi*8+0..7,
// halves 8-15 -> K=k0+16+hi*8+0..7
__device__ __forceinline__ v16h ldA(const _Float16* __restrict__ Ap, int lda,
                                    int k0, int hi, int mt) {
    const _Float16* p = Ap + (size_t)(16 * mt) * lda + k0 + hi * 8;
    return cat16(*(const h8*)p, *(const h8*)(p + 16));
}

#if USE_TDM
// Issue a 2D Tensor-Data-Mover load: nt rows x 32 floats, row stride 64 floats,
// from `src` (global) into `dst` (LDS). D# packing per CDNA5 ISA §8.3/8.4.
__device__ __forceinline__ void tdm_load_bc(const float* src, float* dst, int nt)
{
    unsigned long long ga = (unsigned long long)(const void*)src;
    unsigned int lds = (unsigned int)(unsigned long long)(const void*)dst;
    u32x4 g0;
    g0[0] = 1u;                                        // count=1, user flags 0
    g0[1] = lds;                                       // lds_addr (bytes)
    g0[2] = (unsigned int)(ga & 0xFFFFFFFFu);          // global_addr[31:0]
    g0[3] = (unsigned int)((ga >> 32) & 0x01FFFFFFu)   // global_addr[56:32]
            | (2u << 30);                              // type = 2 (image)
    i32x8v g1;
    g1[0] = (2 << 16);           // data_size = 2 (4 bytes)
    g1[1] = (32 << 16);          // tensor_dim0 = 32 (bits 63:48)
    g1[2] = (nt & 0xFFFF) << 16; // tensor_dim0_hi=0 | tensor_dim1 = nt
    g1[3] = (32 << 16);          // tensor_dim1_hi=0 | tile_dim0 = 32
    g1[4] = nt;                  // tile_dim1 = nt, tile_dim2 = 0
    g1[5] = 64;                  // tensor_dim0_stride = 64 elements
    g1[6] = 0;
    g1[7] = 0;
    i32x4v z4 = {0, 0, 0, 0};    // groups 2/3 unused (2D tensor)
#if __clang_major__ >= 23
    i32x8v z8 = {0, 0, 0, 0, 0, 0, 0, 0};
    __builtin_amdgcn_tensor_load_to_lds(g0, g1, z4, z4, z8, 0);
#else
    __builtin_amdgcn_tensor_load_to_lds(g0, g1, z4, z4, 0);
#endif
}
#endif

// ---------------------------------------------------------------------------
// Guard-free, software-pipelined WMMA GEMM on f16 operands:
//   C[m,n] = act( sum_k A[m,k]*B[n,k] + bias[n] )
// block=(32,4): 4 waves; wave owns one 16-col N tile x 4 stacked 16-row M tiles.
// Next K-step's operands are loaded before the current step's WMMAs issue, so
// WMMAs only wait on loads from the previous iteration (partial loadcnt waits).
// ---------------------------------------------------------------------------
__global__ void wmma_gemm_f16(const _Float16* __restrict__ A, int lda,
                              const _Float16* __restrict__ B, int ldb,
                              float* __restrict__ C, _Float16* __restrict__ C16,
                              int ldc, int K,
                              const float* __restrict__ bias, int act)
{
    int lane = threadIdx.x;
    int hi   = lane >> 4;
    int l16  = lane & 15;
    int tileN = blockIdx.x * 4 + threadIdx.y;
    int ncol  = tileN * 16 + l16;
    int mrow  = blockIdx.y * 64 + l16;

    const _Float16* Bp = B + (size_t)ncol * ldb;
    const _Float16* Ap = A + (size_t)mrow * lda;

    v8f acc[4] = {{}, {}, {}, {}};
    v16h bCur = ldB(Bp, 0, hi);
    v16h aCur[4];
#pragma unroll
    for (int mt = 0; mt < 4; ++mt) aCur[mt] = ldA(Ap, lda, 0, hi, mt);

    for (int k0 = 32; k0 < K; k0 += 32) {
        v16h bNxt = ldB(Bp, k0, hi);
        v16h aNxt[4];
#pragma unroll
        for (int mt = 0; mt < 4; ++mt) aNxt[mt] = ldA(Ap, lda, k0, hi, mt);
#pragma unroll
        for (int mt = 0; mt < 4; ++mt)
            acc[mt] = __builtin_amdgcn_wmma_f32_16x16x32_f16(
                          false, aCur[mt], false, bCur, (short)0, acc[mt], false, false);
        bCur = bNxt;
#pragma unroll
        for (int mt = 0; mt < 4; ++mt) aCur[mt] = aNxt[mt];
    }
#pragma unroll
    for (int mt = 0; mt < 4; ++mt)
        acc[mt] = __builtin_amdgcn_wmma_f32_16x16x32_f16(
                      false, aCur[mt], false, bCur, (short)0, acc[mt], false, false);

    float bv = bias ? bias[ncol] : 0.f;
#pragma unroll
    for (int mt = 0; mt < 4; ++mt) {
#pragma unroll
        for (int r = 0; r < 8; ++r) {
            int m = blockIdx.y * 64 + mt * 16 + r + hi * 8;
            float v = acc[mt][r] + bv;
            if (act == 1) v = dev_softplus(v);
            size_t idx = (size_t)m * ldc + ncol;
            C[idx] = v;
            if (C16) C16[idx] = (_Float16)v;
        }
    }
}

// ---------------------------------------------------------------------------
// weight prep: fp32 -> fp16 (vectorized x4)
__global__ void cvt_f16_v4(const float* __restrict__ in, _Float16* __restrict__ out, int n4)
{
    int i = blockIdx.x * 256 + threadIdx.x;
    if (i < n4) {
        f4 v = ((const f4*)in)[i];
        h4 o;
        o[0] = (_Float16)v[0]; o[1] = (_Float16)v[1];
        o[2] = (_Float16)v[2]; o[3] = (_Float16)v[3];
        ((h4*)out)[i] = o;
    }
}

// x_proj weights (DEPTH x 56 x 768) -> padded/reordered f16 (DEPTH x 64 x 768):
// new row r: r<24 -> old r (dt); 24<=r<32 -> 0; r>=32 -> old r-8 (B then C)
__global__ void prep_xpw(const float* __restrict__ w, _Float16* __restrict__ outp)
{
    int i = blockIdx.x * 256 + threadIdx.x;
    if (i >= DEPTH * DXPP * DI) return;
    int l = i / (DXPP * DI), rr = i % (DXPP * DI);
    int r = rr / DI, k = rr % DI;
    float v = 0.f;
    if (r < 24)       v = w[((size_t)l * 56 + r) * DI + k];
    else if (r >= 32) v = w[((size_t)l * 56 + (r - 8)) * DI + k];
    outp[i] = (_Float16)v;
}

// dt_proj weights (DEPTH x 768 x 24) -> K-padded f16 (DEPTH x 768 x 32)
__global__ void prep_dtw(const float* __restrict__ w, _Float16* __restrict__ outp)
{
    int i = blockIdx.x * 256 + threadIdx.x;
    if (i >= DEPTH * DI * KDT) return;
    int l = i / (DI * KDT), rr = i % (DI * KDT);
    int r = rr / KDT, k = rr % KDT;
    float v = (k < DTR) ? w[((size_t)l * DI + r) * DTR + k] : 0.f;
    outp[i] = (_Float16)v;
}

// ---------------------------------------------------------------------------
// im2col for patch conv, emitted directly as f16. patch_w is (384 x 768) = B.
__global__ void im2col_patch(const float* __restrict__ x, _Float16* __restrict__ Acol)
{
    int id = blockIdx.x * blockDim.x + threadIdx.x;
    if (id >= PM * DI) return;
    int m = id / DI, k = id % DI;
    int t = m / NPATCH, pp = m % NPATCH;
    int hh = pp / 14, ww = pp % 14;
    int ci = k / 256, rr = k % 256;
    int kh = rr / 16, kw = rr % 16;
    Acol[id] = (_Float16)x[((size_t)(ci * TSEQ + t) * 224 + (hh * 16 + kh)) * 224
                           + (ww * 16 + kw)];
}

// Token assembly (also zero-inits residual)
__global__ void build_tokens(const float* __restrict__ P,
                             const float* __restrict__ cls, const float* __restrict__ pe,
                             const float* __restrict__ tp,  const float* __restrict__ ie,
                             float* __restrict__ hidden, float* __restrict__ residual)
{
    int id = blockIdx.x * blockDim.x + threadIdx.x;
    if (id >= L_TOK * DM) return;
    int tok = id / DM, c = id % DM;
    float v;
    if (tok == 0) {
        v = cls[c] + pe[c];
    } else {
        int r = tok - 1;
        int t = r / 392, j = r % 392;
        int s = j / NPATCH, n = j % NPATCH;
        if (s == 0) v = P[(size_t)(t * NPATCH + n) * DM + c] + pe[(size_t)(n + 1) * DM + c];
        else        v = ie[t * DM + c];
        v += tp[t * DM + c];
    }
    hidden[id]   = v;
    residual[id] = 0.f;
}

// residual = hidden + residual; hnorm16 = f16(rms(residual) * nw)
__global__ void add_rms(const float* __restrict__ hidden, float* __restrict__ residual,
                        const float* __restrict__ nw, _Float16* __restrict__ hout16)
{
    int tok = blockIdx.x;
    int t = threadIdx.x;                 // 128 threads, 3 elems each
    __shared__ float red[128];
    float vloc[3];
    float ss = 0.f;
#pragma unroll
    for (int i = 0; i < 3; ++i) {
        int c = t + i * 128;
        float v = hidden[(size_t)tok * DM + c] + residual[(size_t)tok * DM + c];
        vloc[i] = v;
        ss += v * v;
    }
    red[t] = ss; __syncthreads();
    for (int s = 64; s > 0; s >>= 1) { if (t < s) red[t] += red[t + s]; __syncthreads(); }
    float scale = rsqrtf(red[0] / (float)DM + 1e-5f);
#pragma unroll
    for (int i = 0; i < 3; ++i) {
        int c = t + i * 128;
        residual[(size_t)tok * DM + c] = vloc[i];
        hout16[(size_t)tok * DM + c]   = (_Float16)(vloc[i] * scale * nw[c]);
    }
}

// causal depthwise conv (width 4) + silu; dir=1 reads the sequence flipped.
__global__ void conv_silu(const float* __restrict__ xz, const float* __restrict__ cw,
                          const float* __restrict__ cb, float* __restrict__ xc,
                          _Float16* __restrict__ xc16, int dir)
{
    int id = blockIdx.x * blockDim.x + threadIdx.x;
    if (id >= L_TOK * DI) return;
    int tt = id / DI, d = id % DI;
    float acc = cb[d];
#pragma unroll
    for (int j = 0; j < 4; ++j) {
        int tj = tt - 3 + j;
        if (tj >= 0) {
            int tok = dir ? (L_TOK - 1 - tj) : tj;
            acc += xz[(size_t)tok * (2 * DI) + d] * cw[d * 4 + j];
        }
    }
    float v = dev_silu(acc);
    xc[id]   = v;
    xc16[id] = (_Float16)v;
}

// ---------------------------------------------------------------------------
// Selective scan: one lane per channel, 16 states in registers. The shared
// B/C block (SCH timesteps x 32 floats) is staged into LDS — double-buffered
// via the Tensor Data Mover when available: chunk t+1's DMA is issued before
// computing chunk t, and only waited on (s_wait_tensorcnt 0 + barrier) at the
// chunk boundary, overlapping the DMA with the recurrence. Fallback is a
// cooperative copy loop.
// dbl row layout (ld=64): [32:48)=B, [48:64)=C. dir=0 stores y; dir=1 reads the
// fwd result, accumulates, and emits the f16 operand for the out_proj GEMM.
// ---------------------------------------------------------------------------
__global__ void ssm_scan(const float* __restrict__ dt,  const float* __restrict__ xc,
                         const float* __restrict__ dbl, const float* __restrict__ alog,
                         const float* __restrict__ dd,  const float* __restrict__ xz,
                         float* __restrict__ ysum, _Float16* __restrict__ ysum16,
                         int dir)
{
    __shared__ float bc[2][SCH * 32];
    int tid = threadIdx.x;               // 128
    int d = blockIdx.x * 128 + tid;      // grid 6 x 128 = 768, no guard
    float a[DS], h[DS];
#pragma unroll
    for (int s = 0; s < DS; ++s) { a[s] = -expf(alog[(size_t)d * DS + s]); h[s] = 0.f; }
    float ddv = dd[d];
    int buf = 0;

#if USE_TDM
    // prologue: DMA chunk 0 into bc[0]
    if (tid < 32) {
        int nt0 = (L_TOK < SCH) ? L_TOK : SCH;
        tdm_load_bc(dbl + 32, &bc[0][0], nt0);
        __builtin_amdgcn_s_wait_tensorcnt(0);
    }
    __syncthreads();
#endif

    for (int t0 = 0; t0 < L_TOK; t0 += SCH) {
        int nt = (L_TOK - t0 < SCH) ? (L_TOK - t0) : SCH;
#if USE_TDM
        // issue DMA for the NEXT chunk into the other buffer; overlap with compute
        int t1 = t0 + SCH;
        if (t1 < L_TOK && tid < 32) {
            int nt2 = (L_TOK - t1 < SCH) ? (L_TOK - t1) : SCH;
            tdm_load_bc(dbl + (size_t)t1 * DXPP + 32, &bc[1 - buf][0], nt2);
        }
#else
        __syncthreads();
        for (int i = tid; i < nt * 32; i += 128)
            bc[buf][i] = dbl[(size_t)(t0 + (i >> 5)) * DXPP + 32 + (i & 31)];
        __syncthreads();
#endif
        for (int ttc = 0; ttc < nt; ++ttc) {
            int t = t0 + ttc;
            int tok = dir ? (L_TOK - 1 - t) : t;
            float dtv = dt[(size_t)t * DI + d];
            float xcv = xc[(size_t)t * DI + d];
            float du  = dtv * xcv;
            const float* row = &bc[buf][ttc * 32];
            float y = 0.f;
#pragma unroll
            for (int s = 0; s < DS; ++s) {
                float dA = expf(dtv * a[s]);
                h[s] = dA * h[s] + du * row[s];
                y   += h[s] * row[16 + s];
            }
            y += xcv * ddv;
            float zv = xz[(size_t)tok * (2 * DI) + DI + d];
            y *= dev_silu(zv);
            size_t idx = (size_t)tok * DI + d;
            if (dir) {
                float tot = ysum[idx] + y;   // fwd kernel completed earlier (stream order)
                ysum16[idx] = (_Float16)tot;
            } else {
                ysum[idx] = y;
            }
        }
#if USE_TDM
        if (tid < 32) __builtin_amdgcn_s_wait_tensorcnt(0);
        __syncthreads();                 // next chunk's LDS now valid for all waves
        buf ^= 1;
#endif
    }
}

// Final RMS (token 0 only) + classification head (1000 x 384)
__global__ void final_head(const float* __restrict__ hidden, const float* __restrict__ residual,
                           const float* __restrict__ nfw, const float* __restrict__ hw,
                           const float* __restrict__ hb, float* __restrict__ out)
{
    __shared__ float f[DM];
    __shared__ float red[256];
    int t = threadIdx.x; // 256
    float ss = 0.f;
    for (int c = t; c < DM; c += 256) {
        float v = hidden[c] + residual[c];
        f[c] = v;
        ss += v * v;
    }
    red[t] = ss; __syncthreads();
    for (int s = 128; s > 0; s >>= 1) { if (t < s) red[t] += red[t + s]; __syncthreads(); }
    float scale = rsqrtf(red[0] / (float)DM + 1e-5f);
    __syncthreads();
    for (int c = t; c < DM; c += 256) f[c] = f[c] * scale * nfw[c];
    __syncthreads();
    for (int j = t; j < 1000; j += 256) {
        float acc = hb[j];
        for (int c = 0; c < DM; ++c) acc += f[c] * hw[(size_t)j * DM + c];
        out[j] = acc;
    }
}

// ---------------------------------------------------------------------------
extern "C" void kernel_launch(void* const* d_in, const int* in_sizes, int n_in,
                              void* d_out, int out_size, void* d_ws, size_t ws_size,
                              hipStream_t stream)
{
    const float* x    = (const float*)d_in[0];
    const float* ie   = (const float*)d_in[1];
    const float* pw   = (const float*)d_in[2];
    const float* pb   = (const float*)d_in[3];
    const float* ct   = (const float*)d_in[4];
    const float* pe   = (const float*)d_in[5];
    const float* tp   = (const float*)d_in[6];
    const float* nw   = (const float*)d_in[7];
    const float* ipw  = (const float*)d_in[8];
    const float* cw   = (const float*)d_in[9];
    const float* cb   = (const float*)d_in[10];
    const float* xpw  = (const float*)d_in[11];
    const float* dtw  = (const float*)d_in[12];
    const float* dtb  = (const float*)d_in[13];
    const float* alog = (const float*)d_in[14];
    const float* dd   = (const float*)d_in[15];
    const float* cwb  = (const float*)d_in[16];
    const float* cbb  = (const float*)d_in[17];
    const float* xpwb = (const float*)d_in[18];
    const float* dtwb = (const float*)d_in[19];
    const float* dtbb = (const float*)d_in[20];
    const float* alogb= (const float*)d_in[21];
    const float* ddb  = (const float*)d_in[22];
    const float* opw  = (const float*)d_in[23];
    const float* nfw  = (const float*)d_in[24];
    const float* hw   = (const float*)d_in[25];
    const float* hb   = (const float*)d_in[26];
    float* out = (float*)d_out;
    (void)in_sizes; (void)n_in; (void)out_size; (void)ws_size;

    // ---- workspace carve-up: fp32 region then f16 region (~100 MB total) ----
    float* wf = (float*)d_ws;
    size_t off = 0;
    auto af = [&](size_t n) { float* p = wf + off; off += (n + 7) & ~(size_t)7; return p; };
    float* P      = af((size_t)PM_PAD * DM);
    float* hidden = af((size_t)L_PAD * DM);
    float* resid  = af((size_t)L_PAD * DM);
    float* xzb    = af((size_t)L_PAD * 2 * DI);
    float* xcf    = af((size_t)L_PAD * DI);
    float* xcb    = af((size_t)L_PAD * DI);
    float* dblf   = af((size_t)L_PAD * DXPP);
    float* dblb   = af((size_t)L_PAD * DXPP);
    float* dtf    = af((size_t)L_PAD * DI);
    float* dtbf   = af((size_t)L_PAD * DI);
    float* ysum   = af((size_t)L_PAD * DI);

    _Float16* wh = (_Float16*)(wf + off);
    size_t off16 = 0;
    auto ah = [&](size_t n) { _Float16* p = wh + off16; off16 += (n + 15) & ~(size_t)15; return p; };
    _Float16* pw16    = ah((size_t)DM * DI);
    _Float16* ipw16   = ah((size_t)DEPTH * 2 * DI * DM);
    _Float16* opw16   = ah((size_t)DEPTH * DM * DI);
    _Float16* xpw16f  = ah((size_t)DEPTH * DXPP * DI);
    _Float16* xpw16b  = ah((size_t)DEPTH * DXPP * DI);
    _Float16* dtw16f  = ah((size_t)DEPTH * DI * KDT);
    _Float16* dtw16b  = ah((size_t)DEPTH * DI * KDT);
    _Float16* Acol16  = ah((size_t)PM_PAD * DI);
    _Float16* hn16    = ah((size_t)L_PAD * DM);
    _Float16* xcf16   = ah((size_t)L_PAD * DI);
    _Float16* xcb16   = ah((size_t)L_PAD * DI);
    _Float16* dblf16  = ah((size_t)L_PAD * DXPP);
    _Float16* dblb16  = ah((size_t)L_PAD * DXPP);
    _Float16* ysum16  = ah((size_t)L_PAD * DI);

    auto cvt = [&](const float* src, _Float16* dst, size_t n) {
        int n4 = (int)(n / 4);
        cvt_f16_v4<<<(n4 + 255) / 256, 256, 0, stream>>>(src, dst, n4);
    };
    auto gemm = [&](const _Float16* A, int lda, const _Float16* B, int ldb,
                    float* C, _Float16* C16, int ldc, int Mrows, int N, int K,
                    const float* bias, int act) {
        dim3 grid(N / 64, Mrows / 64);   // N, Mrows multiples of 64
        dim3 block(32, 4);
        wmma_gemm_f16<<<grid, block, 0, stream>>>(A, lda, B, ldb, C, C16, ldc, K, bias, act);
    };

    // ---- once-per-call weight prep (f16 + padding/reorder) ----
    cvt(pw,  pw16,  (size_t)DM * DI);
    cvt(ipw, ipw16, (size_t)DEPTH * 2 * DI * DM);
    cvt(opw, opw16, (size_t)DEPTH * DM * DI);
    prep_xpw<<<(DEPTH * DXPP * DI + 255) / 256, 256, 0, stream>>>(xpw,  xpw16f);
    prep_xpw<<<(DEPTH * DXPP * DI + 255) / 256, 256, 0, stream>>>(xpwb, xpw16b);
    prep_dtw<<<(DEPTH * DI * KDT + 255) / 256, 256, 0, stream>>>(dtw,  dtw16f);
    prep_dtw<<<(DEPTH * DI * KDT + 255) / 256, 256, 0, stream>>>(dtwb, dtw16b);

    // ---- patch embedding ----
    im2col_patch<<<(PM * DI + 255) / 256, 256, 0, stream>>>(x, Acol16);
    gemm(Acol16, DI, pw16, DI, P, nullptr, DM, PM_PAD, DM, DI, pb, 0);
    build_tokens<<<(L_TOK * DM + 255) / 256, 256, 0, stream>>>(P, ct, pe, tp, ie, hidden, resid);

    // ---- 24 bidirectional Mamba blocks ----
    for (int l = 0; l < DEPTH; ++l) {
        add_rms<<<L_TOK, 128, 0, stream>>>(hidden, resid, nw + (size_t)l * DM, hn16);

        // xz = h @ in_proj.T  (L x 1536)
        gemm(hn16, DM, ipw16 + (size_t)l * 2 * DI * DM, DM,
             xzb, nullptr, 2 * DI, L_PAD, 2 * DI, DM, nullptr, 0);

        // depthwise causal conv + silu (fwd / flipped bwd)
        conv_silu<<<(L_TOK * DI + 255) / 256, 256, 0, stream>>>(
            xzb, cw  + (size_t)l * DI * 4, cb  + (size_t)l * DI, xcf, xcf16, 0);
        conv_silu<<<(L_TOK * DI + 255) / 256, 256, 0, stream>>>(
            xzb, cwb + (size_t)l * DI * 4, cbb + (size_t)l * DI, xcb, xcb16, 1);

        // x_proj: (L,768) -> (L,64)  [dt|0|B|C]
        gemm(xcf16, DI, xpw16f + (size_t)l * DXPP * DI, DI,
             dblf, dblf16, DXPP, L_PAD, DXPP, DI, nullptr, 0);
        gemm(xcb16, DI, xpw16b + (size_t)l * DXPP * DI, DI,
             dblb, dblb16, DXPP, L_PAD, DXPP, DI, nullptr, 0);

        // dt = softplus(dbl[:, :32] @ dtw_pad.T + dtb): (L,32)->(L,768)
        gemm(dblf16, DXPP, dtw16f + (size_t)l * DI * KDT, KDT,
             dtf,  nullptr, DI, L_PAD, DI, KDT, dtb  + (size_t)l * DI, 1);
        gemm(dblb16, DXPP, dtw16b + (size_t)l * DI * KDT, KDT,
             dtbf, nullptr, DI, L_PAD, DI, KDT, dtbb + (size_t)l * DI, 1);

        // selective scan: fwd writes ysum (f32); bwd accumulates + emits f16
        ssm_scan<<<DI / 128, 128, 0, stream>>>(
            dtf,  xcf, dblf, alog  + (size_t)l * DI * DS, dd  + (size_t)l * DI,
            xzb, ysum, ysum16, 0);
        ssm_scan<<<DI / 128, 128, 0, stream>>>(
            dtbf, xcb, dblb, alogb + (size_t)l * DI * DS, ddb + (size_t)l * DI,
            xzb, ysum, ysum16, 1);

        // hidden = (yf+yb) @ out_proj.T : (L,768)->(L,384)
        gemm(ysum16, DI, opw16 + (size_t)l * DM * DI, DI,
             hidden, nullptr, DM, L_PAD, DM, DI, nullptr, 0);
    }

    // ---- final norm (token 0) + head ----
    final_head<<<1, 256, 0, stream>>>(hidden, resid, nfw, hw, hb, out);
}
